// GAT_23553600651538
// MI455X (gfx1250) — matrix operand order repaired and checked by reading, not compiled, not run on hardware
//
#include <hip/hip_runtime.h>
#include <hip/hip_bf16.h>
#include <math.h>

typedef __attribute__((ext_vector_type(16))) _Float16 v16h;
typedef __attribute__((ext_vector_type(8)))  _Float16 v8h;
typedef __attribute__((ext_vector_type(8)))  float    v8f;

#define GAT_NEG_SLOPE 0.2f

// ---------------------------------------------------------------------------
// float atomic max via signed-max / unsigned-min trick (monotone IEEE order)
// ---------------------------------------------------------------------------
__device__ __forceinline__ void atomicMaxF(float* addr, float val) {
  if (val >= 0.0f) atomicMax((int*)addr, __float_as_int(val));
  else             atomicMin((unsigned int*)addr, __float_as_uint(val));
}

// ---------------------------------------------------------------------------
// WMMA GEMM:  C[M,N] = A[M,K] @ B[K,N], A f16 row-major [M,K], B given as
// f16 pre-transposed Bt[N,K].  fp32 accum/out.  K compile-time (full unroll),
// N multiple of 64.  block = 128 threads (4 waves), 64x64 tile per block.
// Tiles staged with GLOBAL_LOAD_ASYNC_TO_LDS_B128 (ASYNCcnt), double-buffered
// so the next tile's HBM->LDS DMA overlaps the current tile's WMMAs.
// ---------------------------------------------------------------------------
template <int K>
__global__ __launch_bounds__(128)
void gat_gemm_f16(const _Float16* __restrict__ A, const _Float16* __restrict__ Bt,
                  float* __restrict__ C, int M, int N)
{
  __shared__ __align__(16) _Float16 As[2][64][40];   // [buf][row][k], +8 pad
  __shared__ __align__(16) _Float16 Bs[2][64][40];   // [buf][n][k],   +8 pad

  const int tid    = threadIdx.x;
  const int lane   = tid & 31;
  const int wave   = tid >> 5;        // 0..3
  const int halfId = lane >> 4;       // 0/1
  const int l15    = lane & 15;

  const int row0 = blockIdx.y * 64;
  const int n0   = blockIdx.x * 64;

  // issue one tile's async HBM->LDS copies: 4 instructions per wave
  // (2 x 16B chunks of A + 2 x 16B chunks of Bt per thread).
  auto issue_tile = [&](int k0, int buf) {
#pragma unroll
    for (int i = 0; i < 2; ++i) {
      const int l   = tid + 128 * i;     // 0..255 chunk id
      const int r   = l >> 2;            // tile row / tile n
      const int seg = (l & 3) << 3;      // half-offset within row (0,8,16,24)
      int gr = row0 + r; if (gr >= M) gr = M - 1;   // clamp: never stored
      const _Float16* ga = &A [(size_t)gr       * K + k0 + seg];
      const _Float16* gb = &Bt[(size_t)(n0 + r) * K + k0 + seg];
      // low 32 bits of the generic pointer == LDS byte address (flat aperture)
      unsigned la = (unsigned)(size_t)&As[buf][r][seg];
      unsigned lb = (unsigned)(size_t)&Bs[buf][r][seg];
      asm volatile("global_load_async_to_lds_b128 %0, %1, off"
                   :: "v"(la), "v"(ga) : "memory");
      asm volatile("global_load_async_to_lds_b128 %0, %1, off"
                   :: "v"(lb), "v"(gb) : "memory");
    }
  };

  v8f acc[4];
#pragma unroll
  for (int i = 0; i < 4; ++i) acc[i] = (v8f){0.f,0.f,0.f,0.f,0.f,0.f,0.f,0.f};

  constexpr int STEPS = K / 32;
  issue_tile(0, 0);                      // prologue: tile 0 in flight

#pragma unroll
  for (int s = 0; s < STEPS; ++s) {
    const int buf = s & 1;
    if (s + 1 < STEPS) {
      issue_tile((s + 1) * 32, buf ^ 1);           // prefetch next tile
      // 8 outstanding, in-order: <=4 left means tile s has fully landed
      asm volatile("s_wait_asynccnt 0x4" ::: "memory");
    } else {
      asm volatile("s_wait_asynccnt 0x0" ::: "memory");
    }
    __syncthreads();   // tile s visible to all waves; prev buf free to reuse

    // A fragment (16x32 f16): lanes 0-15 K{0-7,16-23}, lanes 16-31 K{8-15,24-31}
    const int arow = wave * 16 + l15;
    v8h a0 = *(const v8h*)&As[buf][arow][halfId * 8];
    v8h a1 = *(const v8h*)&As[buf][arow][16 + halfId * 8];
    v16h afrag;
#pragma unroll
    for (int i = 0; i < 8; ++i) { afrag[i] = a0[i]; afrag[8 + i] = a1[i]; }

#pragma unroll
    for (int ns = 0; ns < 4; ++ns) {
      const int bn = ns * 16 + l15;
      // B fragment (32x16 f16): lanes 0-15 K{0-7,16-23}, lanes 16-31 K{8-15,24-31}
      v8h b0v = *(const v8h*)&Bs[buf][bn][halfId * 8];
      v8h b1v = *(const v8h*)&Bs[buf][bn][16 + halfId * 8];
      v16h bfrag;
#pragma unroll
      for (int i = 0; i < 8; ++i) { bfrag[i] = b0v[i]; bfrag[8 + i] = b1v[i]; }
      acc[ns] = __builtin_amdgcn_wmma_f32_16x16x32_f16(
          false, afrag, false, bfrag, (short)0, acc[ns], false, false);
    }
    __syncthreads();   // all waves done reading buf before iter s+1 overwrites
  }

  // store: C/D layout -> row = j + 8*halfId, col = lane&15
#pragma unroll
  for (int ns = 0; ns < 4; ++ns) {
    const int col = n0 + ns * 16 + l15;
#pragma unroll
    for (int j = 0; j < 8; ++j) {
      const int row = row0 + wave * 16 + j + 8 * halfId;
      if (row < M) C[(size_t)row * N + col] = acc[ns][j];
    }
  }
}

// ---------------------------------------------------------------------------
// precision / layout prep kernels
// ---------------------------------------------------------------------------
__global__ void gat_convert_f16(const float* __restrict__ x, _Float16* __restrict__ y,
                                size_t n) {
  size_t i = (size_t)blockIdx.x * blockDim.x + threadIdx.x;
  if (i < n) y[i] = (_Float16)x[i];
}

// Wt[n,k] = (f16) W[k,n]   (W is [K,N] row-major)
__global__ void gat_transpose_convert(const float* __restrict__ W, _Float16* __restrict__ Wt,
                                      int K, int N) {
  int i = blockIdx.x * blockDim.x + threadIdx.x;
  if (i >= N * K) return;
  int n = i / K, k = i - n * K;
  Wt[i] = (_Float16)W[(size_t)k * N + n];
}

// y = (f16) relu(x + b[col])  -- fused layer-0 epilogue + layer-1 input prep
__global__ void gat_bias_relu_f16(const float* __restrict__ x, const float* __restrict__ b,
                                  _Float16* __restrict__ y, size_t total, int C) {
  size_t i = (size_t)blockIdx.x * blockDim.x + threadIdx.x;
  if (i >= total) return;
  int col = (int)(i % (size_t)C);
  y[i] = (_Float16)fmaxf(x[i] + b[col], 0.f);
}

// ---------------------------------------------------------------------------
// small helper kernels
// ---------------------------------------------------------------------------
__global__ void gat_zero(float* p, size_t n) {
  size_t i = (size_t)blockIdx.x * blockDim.x + threadIdx.x;
  if (i < n) p[i] = 0.0f;
}

__global__ void gat_init_ms(float* m, float* s, int n) {
  int i = blockIdx.x * blockDim.x + threadIdx.x;
  if (i < n) { m[i] = -INFINITY; s[i] = 0.0f; }
}

// per-node attention logits: el[n,h] = <feat[n,h,:], a_s[h,:]>, same for er
__global__ void gat_node_scores(const float* __restrict__ feat,
                                const float* __restrict__ a_s,
                                const float* __restrict__ a_d,
                                float* __restrict__ el, float* __restrict__ er,
                                int H, int D)
{
  extern __shared__ float sh[];
  const int HD = H * D;
  float* ss = sh;
  float* sd = sh + HD;
  const int n = blockIdx.x, t = threadIdx.x;
  const float f = feat[(size_t)n * HD + t];
  ss[t] = f * a_s[t];
  sd[t] = f * a_d[t];
  __syncthreads();
  const int lih = t & (D - 1);
  for (int str = D >> 1; str > 0; str >>= 1) {
    if (lih < str) { ss[t] += ss[t + str]; sd[t] += sd[t + str]; }
    __syncthreads();
  }
  if (lih == 0) {
    const int h = t / D;
    el[n * H + h] = ss[t];
    er[n * H + h] = sd[t];
  }
}

// e = leaky_relu(el[src]+er[dst]); running segment max into m[dst]
__global__ void gat_edge_logits(const int* __restrict__ src, const int* __restrict__ dst,
                                const float* __restrict__ el, const float* __restrict__ er,
                                float* __restrict__ e, float* __restrict__ m,
                                int E, int H)
{
  const int i = blockIdx.x * blockDim.x + threadIdx.x;
  if (i >= E * H) return;
  const int ed = i / H, h = i - ed * H;
  const int sN = src[ed], dN = dst[ed];
  float v = el[sN * H + h] + er[dN * H + h];
  v = (v > 0.f) ? v : GAT_NEG_SLOPE * v;
  e[i] = v;
  atomicMaxF(&m[dN * H + h], v);
}

// e <- exp(e - m[dst]); s[dst] += e
__global__ void gat_edge_exp(const int* __restrict__ dst,
                             float* __restrict__ e, const float* __restrict__ m,
                             float* __restrict__ s, int E, int H)
{
  const int i = blockIdx.x * blockDim.x + threadIdx.x;
  if (i >= E * H) return;
  const int ed = i / H, h = i - ed * H;
  const int dN = dst[ed];
  const float v = expf(e[i] - m[dN * H + h]);
  e[i] = v;
  atomicAdd(&s[dN * H + h], v);
}

// out[dst, :] += feat[src, :] * (ex / (s[dst]+eps)) ; one wave per edge
__global__ void gat_aggregate(const int* __restrict__ src, const int* __restrict__ dst,
                              const float* __restrict__ feat, const float* __restrict__ ex,
                              const float* __restrict__ ssum, float* __restrict__ out,
                              int E, int H, int D)
{
  const int gid  = blockIdx.x * blockDim.x + threadIdx.x;
  const int edge = gid >> 5;
  const int lane = gid & 31;
  if (edge >= E) return;
  const int sN = src[edge], dN = dst[edge];
  const int C = H * D;
  for (int c = lane; c < C; c += 32) {
    const int h = c / D;
    const float alpha = ex[(size_t)edge * H + h] / (ssum[dN * H + h] + 1e-16f);
    atomicAdd(&out[(size_t)dN * C + c], feat[(size_t)sN * C + c] * alpha);
  }
}

// x += bias (broadcast over cols), fp32 in-place (final layer, no relu)
__global__ void gat_bias_add(float* __restrict__ x, const float* __restrict__ b,
                             size_t total, int C)
{
  const size_t i = (size_t)blockIdx.x * blockDim.x + threadIdx.x;
  if (i >= total) return;
  x[i] += b[(int)(i % (size_t)C)];
}

// ---------------------------------------------------------------------------
// host-side launch
// ---------------------------------------------------------------------------
extern "C" void kernel_launch(void* const* d_in, const int* in_sizes, int n_in,
                              void* d_out, int out_size, void* d_ws, size_t ws_size,
                              hipStream_t stream)
{
  const float* x   = (const float*)d_in[0];
  const int*   ei  = (const int*)  d_in[1];
  const float* W0  = (const float*)d_in[2];
  const float* as0 = (const float*)d_in[3];
  const float* ad0 = (const float*)d_in[4];
  const float* b0  = (const float*)d_in[5];
  const float* W1  = (const float*)d_in[6];
  const float* as1 = (const float*)d_in[7];
  const float* ad1 = (const float*)d_in[8];
  const float* b1  = (const float*)d_in[9];

  const int IN_DIM = 256, HID = 64, NHEAD = 4, OUT_DIM = 64, NHEAD_OUT = 1;
  const int C0 = NHEAD * HID;           // 256
  const int C1 = NHEAD_OUT * OUT_DIM;   // 64
  const int N = in_sizes[0] / IN_DIM;   // 50000
  const int E = in_sizes[1] / 2;        // 800000
  const int* src = ei;
  const int* dst = ei + E;

  // workspace arena (byte offsets, 256B aligned); layer 1 reuses buffers
  char* wsb = (char*)d_ws;
  size_t off = 0;
  auto carve = [&](size_t bytes) -> void* {
    void* p = wsb + off;
    off += (bytes + 255) & ~(size_t)255;
    return p;
  };
  float*     feat = (float*)    carve((size_t)N * C0 * sizeof(float));   // GEMM out
  float*     agg  = (float*)    carve((size_t)N * C0 * sizeof(float));   // layer-0 agg
  float*     el   = (float*)    carve((size_t)N * NHEAD * sizeof(float));
  float*     er   = (float*)    carve((size_t)N * NHEAD * sizeof(float));
  float*     mbuf = (float*)    carve((size_t)N * NHEAD * sizeof(float));
  float*     sbuf = (float*)    carve((size_t)N * NHEAD * sizeof(float));
  float*     ebuf = (float*)    carve((size_t)E * NHEAD * sizeof(float));
  _Float16*  ah   = (_Float16*) carve((size_t)N * C0 * sizeof(_Float16)); // f16 A (both layers)
  _Float16*  w0t  = (_Float16*) carve((size_t)C0 * IN_DIM * sizeof(_Float16));
  _Float16*  w1t  = (_Float16*) carve((size_t)C1 * C0 * sizeof(_Float16));
  (void)ws_size; (void)n_in; (void)out_size;

  float* out = (float*)d_out;

  // ======================= layer 0 =======================
  gat_convert_f16<<<(int)(((size_t)N * IN_DIM + 255) / 256), 256, 0, stream>>>(
      x, ah, (size_t)N * IN_DIM);
  gat_transpose_convert<<<(C0 * IN_DIM + 255) / 256, 256, 0, stream>>>(W0, w0t, IN_DIM, C0);
  {
    dim3 gg(C0 / 64, (N + 63) / 64);
    gat_gemm_f16<256><<<gg, 128, 0, stream>>>(ah, w0t, feat, N, C0);
  }
  gat_node_scores<<<N, C0, 2 * C0 * (int)sizeof(float), stream>>>(feat, as0, ad0, el, er, NHEAD, HID);
  gat_init_ms<<<(N * NHEAD + 255) / 256, 256, 0, stream>>>(mbuf, sbuf, N * NHEAD);
  gat_zero<<<(int)(((size_t)N * C0 + 255) / 256), 256, 0, stream>>>(agg, (size_t)N * C0);
  gat_edge_logits<<<(E * NHEAD + 255) / 256, 256, 0, stream>>>(src, dst, el, er, ebuf, mbuf, E, NHEAD);
  gat_edge_exp<<<(E * NHEAD + 255) / 256, 256, 0, stream>>>(dst, ebuf, mbuf, sbuf, E, NHEAD);
  gat_aggregate<<<(int)(((size_t)E * 32 + 255) / 256), 256, 0, stream>>>(
      src, dst, feat, ebuf, sbuf, agg, E, NHEAD, HID);
  // fused: h = relu(agg + b0) converted straight to f16 GEMM-1 input
  gat_bias_relu_f16<<<(int)(((size_t)N * C0 + 255) / 256), 256, 0, stream>>>(
      agg, b0, ah, (size_t)N * C0, C0);

  // ======================= layer 1 =======================
  gat_transpose_convert<<<(C1 * C0 + 255) / 256, 256, 0, stream>>>(W1, w1t, C0, C1);
  {
    dim3 gg(C1 / 64, (N + 63) / 64);
    gat_gemm_f16<256><<<gg, 128, 0, stream>>>(ah, w1t, feat, N, C1);
  }
  gat_node_scores<<<N, C1, 2 * C1 * (int)sizeof(float), stream>>>(feat, as1, ad1, el, er, NHEAD_OUT, OUT_DIM);
  gat_init_ms<<<(N * NHEAD_OUT + 255) / 256, 256, 0, stream>>>(mbuf, sbuf, N * NHEAD_OUT);
  gat_zero<<<(int)(((size_t)N * C1 + 255) / 256), 256, 0, stream>>>(out, (size_t)N * C1);
  gat_edge_logits<<<(E + 255) / 256, 256, 0, stream>>>(src, dst, el, er, ebuf, mbuf, E, 1);
  gat_edge_exp<<<(E + 255) / 256, 256, 0, stream>>>(dst, ebuf, mbuf, sbuf, E, 1);
  gat_aggregate<<<(int)(((size_t)E * 32 + 255) / 256), 256, 0, stream>>>(
      src, dst, feat, ebuf, sbuf, out, E, 1, OUT_DIM);
  gat_bias_add<<<(int)(((size_t)N * C1 + 255) / 256), 256, 0, stream>>>(out, b1, (size_t)N * C1, C1);
}
